// DeepFakeDetectionModel_34162169872474
// MI455X (gfx1250) — matrix operand order repaired and checked by reading, not compiled, not run on hardware
//
#include <hip/hip_runtime.h>
#include <hip/hip_bf16.h>

typedef __bf16 bf16;
typedef __attribute__((ext_vector_type(16))) __bf16 v16bf;
typedef __attribute__((ext_vector_type(8)))  __bf16 v8bf;
typedef __attribute__((ext_vector_type(8)))  float  v8f;
typedef __attribute__((ext_vector_type(4)))  int    v4i;

#define B_    256
#define N_    196
#define C_    768
#define H_    12
#define HD_   64
#define BH_   (B_*H_)      /* 3072  */
#define M1_   (B_*N_)      /* 50176 */
#define N1_   (3*C_)       /* 2304  */
#define EPS_  1e-6f
#define PITCH 40           /* LDS row pitch in bf16 elems (80B: 16B-aligned, bank-spread) */

union FragU { v16bf v; v8bf h[2]; bf16 e[16]; };

__device__ __forceinline__ v8f vzero8() {
  v8f z = {0.f,0.f,0.f,0.f,0.f,0.f,0.f,0.f};
  return z;
}

// ---- gfx1250 async global->LDS path (probe with __has_builtin; sync fallback) ----
#if defined(__has_builtin)
#if __has_builtin(__builtin_amdgcn_global_load_async_to_lds_b128)
#define HAS_ASYNC_LDS 1
#endif
#if __has_builtin(__builtin_amdgcn_s_wait_asynccnt)
#define HAS_WAIT_ASYNC 1
#endif
#endif

__device__ __forceinline__ void stage16(const bf16* g, bf16* l) {
#if defined(HAS_ASYNC_LDS)
  // signature (from clang diagnostic): (v4i addrspace(1)*, v4i addrspace(3)*, imm, imm)
  __builtin_amdgcn_global_load_async_to_lds_b128(
      (__attribute__((address_space(1))) v4i*)g,
      (__attribute__((address_space(3))) v4i*)l, 0, 0);
#else
  *(v8bf*)l = *(const v8bf*)g;   // global load + ds_store fallback
#endif
}

__device__ __forceinline__ void stage_wait() {
#if defined(HAS_WAIT_ASYNC)
  __builtin_amdgcn_s_wait_asynccnt(0);
#elif defined(HAS_ASYNC_LDS)
  asm volatile("s_wait_asynccnt 0x0" ::: "memory");
#endif
}

// ---------------- elementwise f32 -> bf16 ----------------
__global__ void cvt_bf16_kernel(const float* __restrict__ in, bf16* __restrict__ out, int n) {
  int i = blockIdx.x * blockDim.x + threadIdx.x;
  if (i < n) out[i] = (bf16)in[i];
}

// transpose + convert: in [R][Cc] f32 -> out [Cc][R] bf16
__global__ void transpose_bf16_kernel(const float* __restrict__ in, bf16* __restrict__ out, int R, int Cc) {
  int i = blockIdx.x * blockDim.x + threadIdx.x;
  if (i >= R * Cc) return;
  int r = i % R, c = i / R;
  out[i] = (bf16)in[(size_t)r * Cc + c];
}

// ---------------- shared 128x64 block-tile GEMM core, K=768, LDS-staged ----------------
__device__ __forceinline__ void gemm_tile_128x64(
    const bf16* __restrict__ A, const bf16* __restrict__ Bt,
    int mBlock, int nBase, int tid, v8f acc[2][4],
    bf16* sA /*128*PITCH*/, bf16* sB /*64*PITCH*/)
{
  const int lane = tid & 31;
  const int wave = tid >> 5;
  const int l15 = lane & 15, hi = lane >> 4;

#pragma unroll
  for (int i = 0; i < 2; ++i)
#pragma unroll
    for (int j = 0; j < 4; ++j) acc[i][j] = vzero8();

  for (int k0 = 0; k0 < C_; k0 += 32) {
    // ---- stage A tile (128x32) : thread t copies row t (64B = 4 x b128, wave-wide async)
    {
      const bf16* ga = A + (size_t)(mBlock + tid) * C_ + k0;
      bf16* la = sA + tid * PITCH;
#pragma unroll
      for (int c = 0; c < 4; ++c) stage16(ga + c * 8, la + c * 8);
    }
    // ---- stage B tile (64x32) : thread t copies half-row (32B = 2 x b128)
    {
      const bf16* gb = Bt + (size_t)(nBase + (tid >> 1)) * C_ + k0 + (tid & 1) * 16;
      bf16* lb = sB + (tid >> 1) * PITCH + (tid & 1) * 16;
      stage16(gb, lb);
      stage16(gb + 8, lb + 8);
    }
    stage_wait();
    __syncthreads();

    // ---- fragment loads from LDS (ds_load_b128) ----
    v16bf a[2], b[4];
#pragma unroll
    for (int i = 0; i < 2; ++i) {
      // A frag: lane<16 -> K {0..7,16..23}; lane>=16 -> K {8..15,24..31}
      const bf16* p = sA + (wave * 32 + i * 16 + l15) * PITCH + hi * 8;
      FragU u; u.h[0] = *(const v8bf*)p; u.h[1] = *(const v8bf*)(p + 16);
      a[i] = u.v;
    }
#pragma unroll
    for (int j = 0; j < 4; ++j) {
      // B frag: lane<16 -> K {0..7,16..23}; lane>=16 -> K {8..15,24..31}
      const bf16* p = sB + (j * 16 + l15) * PITCH + hi * 8;
      FragU u; u.h[0] = *(const v8bf*)p; u.h[1] = *(const v8bf*)(p + 16);
      b[j] = u.v;
    }
#pragma unroll
    for (int i = 0; i < 2; ++i)
#pragma unroll
      for (int j = 0; j < 4; ++j)
        acc[i][j] = __builtin_amdgcn_wmma_f32_16x16x32_bf16(
            false, a[i], false, b[j], (short)0, acc[i][j], false, false);

    __syncthreads();   // protect LDS tile from next iteration's staging
  }
}

// ---------------- GEMM1: qkv = xb @ w_qkv, fused epilogue to head-major q/k/v ----------------
__global__ void __launch_bounds__(128) gemm_qkv_kernel(
    const bf16* __restrict__ A, const bf16* __restrict__ Bt,
    const float* __restrict__ pos, bf16* __restrict__ qh,
    bf16* __restrict__ kh, bf16* __restrict__ vh)
{
  __shared__ bf16 sA[128 * PITCH];
  __shared__ bf16 sB[64 * PITCH];
  const int tid = threadIdx.x;
  const int lane = tid & 31, wave = tid >> 5;
  const int l15 = lane & 15, hi = lane >> 4;
  const int mBlock = blockIdx.y * 128;
  const int nBase  = blockIdx.x * 64;

  v8f acc[2][4];
  gemm_tile_128x64(A, Bt, mBlock, nBase, tid, acc, sA, sB);

  const int mBase = mBlock + wave * 32;
  const int which = nBase / C_;  // 0=q, 1=k, 2=v (uniform: 768 % 64 == 0)
#pragma unroll
  for (int i = 0; i < 2; ++i)
#pragma unroll
    for (int j = 0; j < 4; ++j)
#pragma unroll
      for (int rr = 0; rr < 8; ++rr) {
        float val = acc[i][j][rr];
        int m = mBase + i * 16 + hi * 8 + rr;     // D tile: M = vgpr + 8*hi
        int n = nBase + j * 16 + l15;             //         N = lane&15
        int tok = m % N_, bb = m / N_;
        int c = n - which * C_;
        int h = c >> 6, d = c & 63;
        size_t hidx = ((size_t)(bb * H_ + h) * N_ + tok) * HD_ + d;
        if (which == 0)      qh[hidx] = (bf16)fmaxf(val, 0.f);
        else if (which == 1) kh[hidx] = (bf16)fmaxf(val + pos[tok * C_ + c], 0.f);
        else                 vh[hidx] = (bf16)val;
      }
}

// ---------------- ksum[bh][d] = sum_j k_h[bh][j][d] ----------------
__global__ void ksum_kernel(const bf16* __restrict__ kh, float* __restrict__ ksum) {
  int i = blockIdx.x * blockDim.x + threadIdx.x;
  if (i >= BH_ * HD_) return;
  int bh = i >> 6, d = i & 63;
  const bf16* p = kh + (size_t)bh * N_ * HD_ + d;
  float s = 0.f;
  for (int j = 0; j < N_; ++j) s += (float)p[j * HD_];
  ksum[i] = s;
}

// ---------------- z[bh][i] = 1/(q_h[bh][i] . ksum[bh] + eps) ----------------
__global__ void zcalc_kernel(const bf16* __restrict__ qh, const float* __restrict__ ksum,
                             float* __restrict__ z) {
  int i = blockIdx.x * blockDim.x + threadIdx.x;
  if (i >= BH_ * N_) return;
  int bh = i / N_, tok = i % N_;
  const bf16*  q = qh + ((size_t)bh * N_ + tok) * HD_;
  const float* s = ksum + (size_t)bh * HD_;
  float acc = 0.f;
#pragma unroll 8
  for (int c = 0; c < HD_; ++c) acc += (float)q[c] * s[c];
  z[i] = 1.f / (acc + EPS_);
}

// ---------------- kv[bh] = k_h^T @ v_h  (64x64, K=196 padded to 224) ----------------
__global__ void __launch_bounds__(128) kv_kernel(const bf16* __restrict__ kh,
                                                 const bf16* __restrict__ vh,
                                                 bf16* __restrict__ kvT) {
  int gw = (blockIdx.x * blockDim.x + threadIdx.x) >> 5;
  if (gw >= BH_) return;
  const int lane = threadIdx.x & 31;
  const int l15 = lane & 15, hi = lane >> 4;
  const bf16* K = kh + (size_t)gw * N_ * HD_;
  const bf16* V = vh + (size_t)gw * N_ * HD_;

  v8f acc[4][4];
#pragma unroll
  for (int ci = 0; ci < 4; ++ci)
#pragma unroll
    for (int di = 0; di < 4; ++di) acc[ci][di] = vzero8();

  for (int j0 = 0; j0 < 224; j0 += 32) {
    v16bf a[4], b[4];
#pragma unroll
    for (int ci = 0; ci < 4; ++ci) {       // A = k^T : A[c][j] = k_h[j][c]
      int c = ci * 16 + l15;
      FragU u;
#pragma unroll
      for (int e = 0; e < 8; ++e) {
        int j1 = j0 + hi * 8 + e;
        int j2 = j0 + 16 + hi * 8 + e;
        u.e[e]     = (j1 < N_) ? K[(size_t)j1 * HD_ + c] : (bf16)0.f;
        u.e[e + 8] = (j2 < N_) ? K[(size_t)j2 * HD_ + c] : (bf16)0.f;
      }
      a[ci] = u.v;
    }
#pragma unroll
    for (int di = 0; di < 4; ++di) {       // B = v_h : B[j][d]
      int d = di * 16 + l15;
      FragU u;
#pragma unroll
      for (int e = 0; e < 16; ++e) {
        int j = j0 + (e & 8) * 2 + hi * 8 + (e & 7);
        u.e[e] = (j < N_) ? V[(size_t)j * HD_ + d] : (bf16)0.f;
      }
      b[di] = u.v;
    }
#pragma unroll
    for (int ci = 0; ci < 4; ++ci)
#pragma unroll
      for (int di = 0; di < 4; ++di)
        acc[ci][di] = __builtin_amdgcn_wmma_f32_16x16x32_bf16(
            false, a[ci], false, b[di], (short)0, acc[ci][di], false, false);
  }

  bf16* out = kvT + (size_t)gw * HD_ * HD_;
#pragma unroll
  for (int ci = 0; ci < 4; ++ci)
#pragma unroll
    for (int di = 0; di < 4; ++di)
#pragma unroll
      for (int rr = 0; rr < 8; ++rr) {
        int c = ci * 16 + hi * 8 + rr;
        int d = di * 16 + l15;
        out[(size_t)d * HD_ + c] = (bf16)acc[ci][di][rr];  // transposed store
      }
}

// ---------------- depthwise 5x5 conv on v (SAME, groups=HD) -> fm[bh][j][d] ----------------
__global__ void conv_kernel(const bf16* __restrict__ vh, const float* __restrict__ w,
                            const float* __restrict__ bias, bf16* __restrict__ fm) {
  int i = blockIdx.x * blockDim.x + threadIdx.x;
  if (i >= BH_ * N_ * HD_) return;
  int d = i & 63;
  int j = (i >> 6) % N_;
  int bh = i / (N_ * HD_);
  int y = j / 14, x = j % 14;
  const bf16* V = vh + (size_t)bh * N_ * HD_;
  const float* wd = w + d * 25;
  float s = bias[d];
#pragma unroll
  for (int ky = 0; ky < 5; ++ky) {
    int yy = y + ky - 2;
    if (yy < 0 || yy >= 14) continue;
#pragma unroll
    for (int kx = 0; kx < 5; ++kx) {
      int xx = x + kx - 2;
      if (xx < 0 || xx >= 14) continue;
      s += (float)V[(size_t)(yy * 14 + xx) * HD_ + d] * wd[ky * 5 + kx];
    }
  }
  fm[i] = (bf16)s;
}

// ---------------- out = (q @ kv) * z + fm -> attn[b][tok][h*64+d] (bf16) ----------------
__global__ void __launch_bounds__(128) out_kernel(const bf16* __restrict__ qh,
                                                  const bf16* __restrict__ kvT,
                                                  const float* __restrict__ z,
                                                  const bf16* __restrict__ fm,
                                                  bf16* __restrict__ attn) {
  int gw = (blockIdx.x * blockDim.x + threadIdx.x) >> 5;
  int bh = gw / 13, mt = gw % 13;
  if (bh >= BH_) return;
  const int lane = threadIdx.x & 31;
  const int l15 = lane & 15, hi = lane >> 4;
  const bf16* Q  = qh  + (size_t)bh * N_ * HD_;
  const bf16* KV = kvT + (size_t)bh * HD_ * HD_;   // [d][c]

  v8f acc[4];
#pragma unroll
  for (int di = 0; di < 4; ++di) acc[di] = vzero8();

  for (int k0 = 0; k0 < HD_; k0 += 32) {
    v16bf a, b[4];
    {
      int row = mt * 16 + l15;
      FragU u;
      if (row < N_) {
        const bf16* p = Q + (size_t)row * HD_ + k0 + hi * 16;
        u.h[0] = *(const v8bf*)p; u.h[1] = *(const v8bf*)(p + 8);
      } else {
#pragma unroll
        for (int e = 0; e < 16; ++e) u.e[e] = (bf16)0.f;
      }
      a = u.v;
    }
#pragma unroll
    for (int di = 0; di < 4; ++di) {
      int d = di * 16 + l15;
      const bf16* p = KV + (size_t)d * HD_ + k0 + hi * 16;  // contiguous K (c)
      FragU u; u.h[0] = *(const v8bf*)p; u.h[1] = *(const v8bf*)(p + 8);
      b[di] = u.v;
    }
#pragma unroll
    for (int di = 0; di < 4; ++di)
      acc[di] = __builtin_amdgcn_wmma_f32_16x16x32_bf16(
          false, a, false, b[di], (short)0, acc[di], false, false);
  }

  int bb = bh / H_, h = bh % H_;
#pragma unroll
  for (int di = 0; di < 4; ++di)
#pragma unroll
    for (int rr = 0; rr < 8; ++rr) {
      int row = mt * 16 + hi * 8 + rr;
      if (row < N_) {
        int d = di * 16 + l15;
        float val = acc[di][rr] * z[(size_t)bh * N_ + row]
                  + (float)fm[((size_t)bh * N_ + row) * HD_ + d];
        attn[((size_t)bb * N_ + row) * C_ + h * HD_ + d] = (bf16)val;
      }
    }
}

// ---------------- GEMM2: y = attn @ w_proj + b_proj (f32 out) ----------------
__global__ void __launch_bounds__(128) gemm_proj_kernel(
    const bf16* __restrict__ A, const bf16* __restrict__ Bt,
    const float* __restrict__ bias, float* __restrict__ out)
{
  __shared__ bf16 sA[128 * PITCH];
  __shared__ bf16 sB[64 * PITCH];
  const int tid = threadIdx.x;
  const int lane = tid & 31, wave = tid >> 5;
  const int l15 = lane & 15, hi = lane >> 4;
  const int mBlock = blockIdx.y * 128;
  const int nBase  = blockIdx.x * 64;

  v8f acc[2][4];
  gemm_tile_128x64(A, Bt, mBlock, nBase, tid, acc, sA, sB);

  const int mBase = mBlock + wave * 32;
#pragma unroll
  for (int i = 0; i < 2; ++i)
#pragma unroll
    for (int j = 0; j < 4; ++j)
#pragma unroll
      for (int rr = 0; rr < 8; ++rr) {
        int m = mBase + i * 16 + hi * 8 + rr;
        int n = nBase + j * 16 + l15;
        out[(size_t)m * C_ + n] = acc[i][j][rr] + bias[n];
      }
}

// ---------------- host orchestration ----------------
extern "C" void kernel_launch(void* const* d_in, const int* in_sizes, int n_in,
                              void* d_out, int out_size, void* d_ws, size_t ws_size,
                              hipStream_t stream) {
  const float* x      = (const float*)d_in[0];  // [256,196,768]
  const float* w_qkv  = (const float*)d_in[1];  // [768,2304]
  const float* pos    = (const float*)d_in[2];  // [1,196,768]
  const float* dwc_w  = (const float*)d_in[3];  // [64,1,5,5]
  const float* dwc_b  = (const float*)d_in[4];  // [64]
  const float* w_proj = (const float*)d_in[5];  // [768,768]
  const float* b_proj = (const float*)d_in[6];  // [768]
  float* y = (float*)d_out;                     // [256,196,768] f32

  char* ws = (char*)d_ws;
  auto alloc = [&](size_t bytes) {
    void* p = (void*)ws;
    ws += (bytes + 255) & ~(size_t)255;
    return p;
  };
  const size_t MC = (size_t)M1_ * C_;
  bf16* xb     = (bf16*)alloc(MC * 2);
  bf16* wqkvT  = (bf16*)alloc((size_t)N1_ * C_ * 2);
  bf16* wprojT = (bf16*)alloc((size_t)C_ * C_ * 2);
  bf16* qh     = (bf16*)alloc((size_t)BH_ * N_ * HD_ * 2);
  bf16* kh     = (bf16*)alloc((size_t)BH_ * N_ * HD_ * 2);
  bf16* vh     = (bf16*)alloc((size_t)BH_ * N_ * HD_ * 2);
  bf16* kvT    = (bf16*)alloc((size_t)BH_ * HD_ * HD_ * 2);
  bf16* fm     = (bf16*)alloc((size_t)BH_ * N_ * HD_ * 2);
  bf16* attn   = (bf16*)alloc(MC * 2);
  float* ksum  = (float*)alloc((size_t)BH_ * HD_ * 4);
  float* zbuf  = (float*)alloc((size_t)BH_ * N_ * 4);

  cvt_bf16_kernel<<<(int)((MC + 255) / 256), 256, 0, stream>>>(x, xb, (int)MC);
  transpose_bf16_kernel<<<(C_ * N1_ + 255) / 256, 256, 0, stream>>>(w_qkv, wqkvT, C_, N1_);
  transpose_bf16_kernel<<<(C_ * C_ + 255) / 256, 256, 0, stream>>>(w_proj, wprojT, C_, C_);
  gemm_qkv_kernel<<<dim3(N1_ / 64, M1_ / 128), 128, 0, stream>>>(xb, wqkvT, pos, qh, kh, vh);
  ksum_kernel<<<(BH_ * HD_ + 255) / 256, 256, 0, stream>>>(kh, ksum);
  zcalc_kernel<<<(BH_ * N_ + 255) / 256, 256, 0, stream>>>(qh, ksum, zbuf);
  kv_kernel<<<(BH_ * 32 + 127) / 128, 128, 0, stream>>>(kh, vh, kvT);
  conv_kernel<<<(int)((MC + 255) / 256), 256, 0, stream>>>(vh, dwc_w, dwc_b, fm);
  out_kernel<<<(BH_ * 13 * 32 + 127) / 128, 128, 0, stream>>>(qh, kvT, zbuf, fm, attn);
  gemm_proj_kernel<<<dim3(C_ / 64, M1_ / 128), 128, 0, stream>>>(attn, wprojT, b_proj, y);
}
